// MultiHeadAttention_25606595019017
// MI455X (gfx1250) — matrix operand order repaired and checked by reading, not compiled, not run on hardware
//
#include <hip/hip_runtime.h>
#include <hip/hip_bf16.h>

typedef __attribute__((ext_vector_type(16))) _Float16 v16h;
typedef __attribute__((ext_vector_type(8)))  _Float16 v8h;
typedef __attribute__((ext_vector_type(8)))  float    v8f;
typedef __attribute__((ext_vector_type(4)))  float    f32x4;

#define HH      8
#define DD      32
#define EE      16
#define NFOCAL  4
#define GHN     (HH - NFOCAL)
#define HIDN    256
#define BB      16
#define NNQ     512
#define SCALEF  0.17677669529663687f   // 1/sqrt(32)

// ---- WMMA register layouts (CDNA5 ISA §7.12.2, wave32) ----
// A 16x32 f16 : lane&15 = row M; this code packs K = hi*16 + 0..15 (contiguous; ISA table order differs)
//               (hi = lane>>4)
// B 32x16 f16 : lane&15 = col N; this code packs K = hi*16 + 0..15 (contiguous; ISA table order differs)
// C/D 16x16 f32: lane&15 = col N; vgpr v = row M = v + 8*hi
// => every operand tile is two contiguous 16-byte runs per lane -> b128 loads.

__device__ __forceinline__ v16h combine(v8h a, v8h b) {
    v16h r;
    #pragma unroll
    for (int t = 0; t < 8; ++t) { r[t] = a[t]; r[8 + t] = b[t]; }
    return r;
}

// ============================================================================
// Kernel 0: vectorized f32 -> f16 conversion (8 elements / thread)
// ============================================================================
__global__ __launch_bounds__(256) void cvt_f16(const float* __restrict__ src,
                                               _Float16* __restrict__ dst, int n8) {
    const int idx = blockIdx.x * 256 + threadIdx.x;
    if (idx >= n8) return;
    f32x4 a = *(const f32x4*)(src + (size_t)idx * 8);
    f32x4 b = *(const f32x4*)(src + (size_t)idx * 8 + 4);
    v8h o;
    #pragma unroll
    for (int t = 0; t < 4; ++t) { o[t] = (_Float16)a[t]; o[4 + t] = (_Float16)b[t]; }
    *(v8h*)(dst + (size_t)idx * 8) = o;
}

// ============================================================================
// Kernel 1: projection GEMM (f16 in, f32 acc).  Wave = 16x64 strip (4 tiles,
// A reused 4x).  transpose_out: 0 -> [B,H,N,D] (q/k), 1 -> [B,H,D,N] (v^T).
// ============================================================================
__global__ __launch_bounds__(256) void proj_gemm(
    const _Float16* __restrict__ X,    // [B*N, HIDN] f16
    const _Float16* __restrict__ W,    // [256, HIDN] f16
    const float* __restrict__ bias,
    _Float16* __restrict__ out,
    int transpose_out)
{
    const int lane = threadIdx.x & 31;
    const int wv   = threadIdx.x >> 5;
    const int wt   = blockIdx.x * 8 + wv;     // 2048 waves
    const int mt   = wt >> 2;                 // 0..511
    const int ng   = wt & 3;                  // column group of 4 tiles
    const int hi = lane >> 4, lo = lane & 15;

    v8f acc[4] = {{}, {}, {}, {}};
    const _Float16* Arow = X + (size_t)(mt * 16 + lo) * HIDN;
    for (int kc = 0; kc < HIDN; kc += 32) {
        const v16h a = combine(*(const v8h*)(Arow + kc + hi * 16),
                               *(const v8h*)(Arow + kc + 8 + hi * 16));
        #pragma unroll
        for (int g = 0; g < 4; ++g) {
            const int n = (ng * 4 + g) * 16 + lo;
            const _Float16* Brow = W + (size_t)n * HIDN + kc + hi * 16;
            const v16h bm = combine(*(const v8h*)(Brow), *(const v8h*)(Brow + 8));
            acc[g] = __builtin_amdgcn_wmma_f32_16x16x32_f16(false, a, false, bm,
                                                            (short)0, acc[g], false, false);
        }
    }
    #pragma unroll
    for (int g = 0; g < 4; ++g) {
        const int col = (ng * 4 + g) * 16 + lo;
        const float bv = bias[col];
        const int h = col >> 5, d = col & 31;
        #pragma unroll
        for (int v = 0; v < 8; ++v) {
            const int row = mt * 16 + v + 8 * hi;
            const int b = row >> 9, n = row & (NNQ - 1);
            const _Float16 val = (_Float16)(acc[g][v] + bv);
            if (transpose_out)
                out[(((size_t)b * HH + h) * DD + d) * NNQ + n] = val;   // [B,H,D,N]
            else
                out[(((size_t)b * HH + h) * NNQ + n) * DD + d] = val;   // [B,H,N,D]
        }
    }
}

// ============================================================================
// Kernel 2: edge-type score table  s[b,h,i,e] = qh.qe_emb + kh.ke_emb
// ============================================================================
__global__ __launch_bounds__(32) void edge_score(
    const _Float16* __restrict__ qh, const _Float16* __restrict__ kh,
    const _Float16* __restrict__ qee, const _Float16* __restrict__ kee, // f16 [E,256]
    float* __restrict__ sbias)
{
    const int lane = threadIdx.x & 31;
    const int hi = lane >> 4, lo = lane & 15;
    const int bh = blockIdx.x >> 5;
    const int it = blockIdx.x & 31;
    const int h  = bh & 7;

    const _Float16* qp = qh + ((size_t)bh * NNQ + it * 16 + lo) * DD;
    const _Float16* kp = kh + ((size_t)bh * NNQ + it * 16 + lo) * DD;
    const v16h qa = combine(*(const v8h*)(qp + hi * 16), *(const v8h*)(qp + 8 + hi * 16));
    const v16h ka = combine(*(const v8h*)(kp + hi * 16), *(const v8h*)(kp + 8 + hi * 16));
    const _Float16* qbp = qee + (size_t)lo * (HH * DD) + h * DD + hi * 16;
    const _Float16* kbp = kee + (size_t)lo * (HH * DD) + h * DD + hi * 16;
    const v16h qb = combine(*(const v8h*)(qbp), *(const v8h*)(qbp + 8));
    const v16h kb = combine(*(const v8h*)(kbp), *(const v8h*)(kbp + 8));

    v8f acc = {};
    acc = __builtin_amdgcn_wmma_f32_16x16x32_f16(false, qa, false, qb, (short)0, acc, false, false);
    acc = __builtin_amdgcn_wmma_f32_16x16x32_f16(false, ka, false, kb, (short)0, acc, false, false);
    #pragma unroll
    for (int v = 0; v < 8; ++v)
        sbias[((size_t)bh * NNQ + it * 16 + v + 8 * hi) * EE + lo] = acc[v];
}

// ============================================================================
// Kernel 3: fused flash attention with edge bias + per-head masking.
// ============================================================================
__global__ __launch_bounds__(128) void attn_kernel(
    const _Float16* __restrict__ qh,  // [B,H,N,D]
    const _Float16* __restrict__ kh,  // [B,H,N,D]
    const _Float16* __restrict__ vT,  // [B,H,D,N]  (transposed for b128 B-tiles)
    const float* __restrict__ sbias,
    const int* __restrict__ edge_attr,
    const unsigned char* __restrict__ mask,
    const unsigned char* __restrict__ fmask,
    _Float16* __restrict__ oh)
{
    __shared__ _Float16 pbuf[4][16 * 32];      // per-wave C->A transpose bounce

    const int lane = threadIdx.x & 31;
    const int wv   = threadIdx.x >> 5;
    const int hi = lane >> 4, lo = lane & 15;
    const int bh = blockIdx.x;
    const int b = bh >> 3, h = bh & 7;
    const int i0 = blockIdx.y * 64 + wv * 16;

    const _Float16* Kb = kh + (size_t)bh * NNQ * DD;
    const _Float16* Vt = vT + (size_t)bh * DD * NNQ;

    const _Float16* qp = qh + ((size_t)bh * NNQ + i0 + lo) * DD;
    const v16h qa = combine(*(const v8h*)(qp + hi * 16), *(const v8h*)(qp + 8 + hi * 16));

    v8f o0 = {}, o1 = {};
    float m[8], l[8];
    #pragma unroll
    for (int v = 0; v < 8; ++v) { m[v] = -3.0e38f; l[v] = 0.f; }

    for (int jb = 0; jb < NNQ; jb += 32) {
        if (jb + 32 < NNQ) {                   // speculative prefetch of next chunk
            __builtin_prefetch(Kb + (size_t)(jb + 32 + lane) * DD, 0, 3);
            __builtin_prefetch(Vt + (size_t)lane * NNQ + jb + 32, 0, 3);
        }
        // --- K^T tiles (B layout: contiguous d-run of 16 per lane) ---
        const _Float16* kp0 = Kb + (size_t)(jb + lo) * DD + hi * 16;
        const _Float16* kp1 = Kb + (size_t)(jb + 16 + lo) * DD + hi * 16;
        const v16h bk0 = combine(*(const v8h*)(kp0), *(const v8h*)(kp0 + 8));
        const v16h bk1 = combine(*(const v8h*)(kp1), *(const v8h*)(kp1 + 8));
        v8f zero = {};
        v8f s0 = __builtin_amdgcn_wmma_f32_16x16x32_f16(false, qa, false, bk0,
                                                        (short)0, zero, false, false);
        v8f s1 = __builtin_amdgcn_wmma_f32_16x16x32_f16(false, qa, false, bk1,
                                                        (short)0, zero, false, false);

        // --- edge bias gather + mask (C layout: row = v+8*hi, col = lo) ---
        float f0[8], f1[8];
        #pragma unroll
        for (int v = 0; v < 8; ++v) {
            const int i  = i0 + v + 8 * hi;
            const int j0 = jb + lo;
            const int j1 = jb + 16 + lo;
            const float* sb = sbias + ((size_t)bh * NNQ + i) * EE;
            const int*   ea = edge_attr + ((size_t)b * NNQ + i) * NNQ;
            const float val0 = (s0[v] + sb[ea[j0]]) * SCALEF;
            const float val1 = (s1[v] + sb[ea[j1]]) * SCALEF;
            bool ok0, ok1;
            if (h < GHN) {
                ok0 = mask[b * NNQ + j0] != 0;
                ok1 = mask[b * NNQ + j1] != 0;
            } else {
                const unsigned char* fm = fmask + ((size_t)b * NNQ + i) * NNQ;
                ok0 = fm[j0] != 0;
                ok1 = fm[j1] != 0;
            }
            f0[v] = ok0 ? val0 : -1.0e30f;
            f1[v] = ok1 ? val1 : -1.0e30f;
        }

        // --- online softmax (row reductions stay inside 16-lane halves) ---
        #pragma unroll
        for (int v = 0; v < 8; ++v) {
            float mt = fmaxf(f0[v], f1[v]);
            #pragma unroll
            for (int off = 1; off < 16; off <<= 1)
                mt = fmaxf(mt, __shfl_xor(mt, off, 32));
            const float mnew = fmaxf(m[v], mt);
            const float corr = __expf(m[v] - mnew);
            const float p0 = __expf(f0[v] - mnew);
            const float p1 = __expf(f1[v] - mnew);
            float rs = p0 + p1;
            #pragma unroll
            for (int off = 1; off < 16; off <<= 1)
                rs += __shfl_xor(rs, off, 32);
            l[v] = l[v] * corr + rs;
            o0[v] *= corr;
            o1[v] *= corr;
            m[v] = mnew;
            pbuf[wv][(v + 8 * hi) * 32 + lo]      = (_Float16)p0;
            pbuf[wv][(v + 8 * hi) * 32 + 16 + lo] = (_Float16)p1;
        }

        // --- P back as A-matrix: two b128 ds loads (same-wave DS is in-order) ---
        const _Float16* pp = &pbuf[wv][lo * 32 + hi * 16];
        const v16h pa = combine(*(const v8h*)(pp), *(const v8h*)(pp + 8));

        // --- V tiles from V^T: contiguous j-run of 16 per lane ---
        const _Float16* vp0 = Vt + (size_t)lo * NNQ + jb + hi * 16;
        const _Float16* vp1 = Vt + (size_t)(16 + lo) * NNQ + jb + hi * 16;
        const v16h bv0 = combine(*(const v8h*)(vp0), *(const v8h*)(vp0 + 8));
        const v16h bv1 = combine(*(const v8h*)(vp1), *(const v8h*)(vp1 + 8));
        o0 = __builtin_amdgcn_wmma_f32_16x16x32_f16(false, pa, false, bv0,
                                                    (short)0, o0, false, false);
        o1 = __builtin_amdgcn_wmma_f32_16x16x32_f16(false, pa, false, bv1,
                                                    (short)0, o1, false, false);
    }

    #pragma unroll
    for (int v = 0; v < 8; ++v) {
        const float inv = l[v] > 0.f ? 1.f / l[v] : 0.f;
        const int i = i0 + v + 8 * hi;
        oh[((size_t)bh * NNQ + i) * DD + lo]      = (_Float16)(o0[v] * inv);
        oh[((size_t)bh * NNQ + i) * DD + 16 + lo] = (_Float16)(o1[v] * inv);
    }
}

// ============================================================================
// Kernel 4: output projection  out = O_flat @ Wo^T + bo   (f32 out)
// ============================================================================
__global__ __launch_bounds__(256) void out_proj(
    const _Float16* __restrict__ oh,   // [B,H,N,D] f16
    const _Float16* __restrict__ Wo,   // [HIDN, 256] f16
    const float* __restrict__ bo,
    float* __restrict__ out)           // [B,N,HIDN]
{
    const int lane = threadIdx.x & 31;
    const int wv   = threadIdx.x >> 5;
    const int wt   = blockIdx.x * 8 + wv;     // 2048 waves
    const int mt   = wt >> 2;
    const int ng   = wt & 3;
    const int hi = lane >> 4, lo = lane & 15;

    const int arow = mt * 16 + lo;
    const int ab = arow >> 9, an = arow & (NNQ - 1);
    const _Float16* Abase = oh + ((size_t)ab * HH * NNQ + an) * DD;  // + h*N*D + d

    v8f acc[4] = {{}, {}, {}, {}};
    for (int kc = 0; kc < 256; kc += 32) {
        const int hh = kc >> 5;                                // head of this K-chunk
        const _Float16* ap = Abase + (size_t)hh * NNQ * DD;
        const v16h a = combine(*(const v8h*)(ap + hi * 16),
                               *(const v8h*)(ap + 8 + hi * 16));
        #pragma unroll
        for (int g = 0; g < 4; ++g) {
            const int c = (ng * 4 + g) * 16 + lo;
            const _Float16* Brow = Wo + (size_t)c * (HH * DD) + kc + hi * 16;
            const v16h bm = combine(*(const v8h*)(Brow), *(const v8h*)(Brow + 8));
            acc[g] = __builtin_amdgcn_wmma_f32_16x16x32_f16(false, a, false, bm,
                                                            (short)0, acc[g], false, false);
        }
    }
    #pragma unroll
    for (int g = 0; g < 4; ++g) {
        const int c = (ng * 4 + g) * 16 + lo;
        const float bv = bo[c];
        #pragma unroll
        for (int v = 0; v < 8; ++v) {
            const int r = mt * 16 + v + 8 * hi;
            out[(size_t)r * HIDN + c] = acc[g][v] + bv;
        }
    }
}

// ============================================================================
extern "C" void kernel_launch(void* const* d_in, const int* in_sizes, int n_in,
                              void* d_out, int out_size, void* d_ws, size_t ws_size,
                              hipStream_t stream) {
    (void)in_sizes; (void)n_in; (void)out_size; (void)ws_size;
    const float* q   = (const float*)d_in[0];
    const float* k   = (const float*)d_in[1];
    const float* v   = (const float*)d_in[2];
    const float* qee = (const float*)d_in[3];
    const float* kee = (const float*)d_in[4];
    const int*   ea  = (const int*)d_in[5];
    const unsigned char* mask  = (const unsigned char*)d_in[6];
    const unsigned char* fmask = (const unsigned char*)d_in[7];
    const float* Wq = (const float*)d_in[8];
    const float* bq = (const float*)d_in[9];
    const float* Wk = (const float*)d_in[10];
    const float* bk = (const float*)d_in[11];
    const float* Wv = (const float*)d_in[12];
    const float* bv = (const float*)d_in[13];
    const float* Wo = (const float*)d_in[14];
    const float* bo = (const float*)d_in[15];
    float* out = (float*)d_out;

    char* ws = (char*)d_ws;
    const size_t MB = 1024 * 1024;
    _Float16* q16   = (_Float16*)(ws);                 // 4 MB each
    _Float16* k16   = (_Float16*)(ws + 4 * MB);
    _Float16* v16   = (_Float16*)(ws + 8 * MB);
    _Float16* wq16  = (_Float16*)(ws + 12 * MB);       // 128 KB each
    _Float16* wk16  = (_Float16*)(ws + 12 * MB + 128 * 1024);
    _Float16* wv16  = (_Float16*)(ws + 12 * MB + 256 * 1024);
    _Float16* wo16  = (_Float16*)(ws + 12 * MB + 384 * 1024);
    _Float16* qee16 = (_Float16*)(ws + 12 * MB + 512 * 1024);  // 8 KB each
    _Float16* kee16 = (_Float16*)(ws + 12 * MB + 520 * 1024);
    _Float16* qhp   = (_Float16*)(ws + 13 * MB);       // [B,H,N,D] 4 MB
    _Float16* khp   = (_Float16*)(ws + 17 * MB);       // [B,H,N,D] 4 MB
    _Float16* vTp   = (_Float16*)(ws + 21 * MB);       // [B,H,D,N] 4 MB
    _Float16* ohp   = (_Float16*)(ws + 25 * MB);       // [B,H,N,D] 4 MB
    float*    sbp   = (float*)   (ws + 29 * MB);       // [B,H,N,E] 4 MB

    const int nAct = BB * NNQ * HIDN;                  // 2,097,152
    const int nW   = HIDN * 256;                       // 65,536
    const int nE   = EE * HH * DD;                     // 4,096
    cvt_f16<<<dim3(nAct / 8 / 256), dim3(256), 0, stream>>>(q, q16, nAct / 8);
    cvt_f16<<<dim3(nAct / 8 / 256), dim3(256), 0, stream>>>(k, k16, nAct / 8);
    cvt_f16<<<dim3(nAct / 8 / 256), dim3(256), 0, stream>>>(v, v16, nAct / 8);
    cvt_f16<<<dim3(nW / 8 / 256),   dim3(256), 0, stream>>>(Wq, wq16, nW / 8);
    cvt_f16<<<dim3(nW / 8 / 256),   dim3(256), 0, stream>>>(Wk, wk16, nW / 8);
    cvt_f16<<<dim3(nW / 8 / 256),   dim3(256), 0, stream>>>(Wv, wv16, nW / 8);
    cvt_f16<<<dim3(nW / 8 / 256),   dim3(256), 0, stream>>>(Wo, wo16, nW / 8);
    cvt_f16<<<dim3(2), dim3(256), 0, stream>>>(qee, qee16, nE / 8);
    cvt_f16<<<dim3(2), dim3(256), 0, stream>>>(kee, kee16, nE / 8);

    const dim3 gblk(256), ggrd(256);                   // 2048 waves: 512 mt x 4 groups
    proj_gemm<<<ggrd, gblk, 0, stream>>>(q16, wq16, bq, qhp, 0);
    proj_gemm<<<ggrd, gblk, 0, stream>>>(k16, wk16, bk, khp, 0);
    proj_gemm<<<ggrd, gblk, 0, stream>>>(v16, wv16, bv, vTp, 1);
    edge_score<<<dim3(BB * HH * (NNQ / 16)), dim3(32), 0, stream>>>(qhp, khp, qee16, kee16, sbp);
    attn_kernel<<<dim3(BB * HH, NNQ / 64), dim3(128), 0, stream>>>(qhp, khp, vTp, sbp,
                                                                   ea, mask, fmask, ohp);
    out_proj<<<ggrd, gblk, 0, stream>>>(ohp, wo16, bo, out);
}
